// FuzzyPDAWithCounter_13022340842132
// MI455X (gfx1250) — matrix operand order repaired and checked by reading, not compiled, not run on hardware
//
#include <hip/hip_runtime.h>

// FuzzyPDAWithCounter for MI455X (gfx1250), compile-only loop.
// Chunked parallel scan (G=1024 chunks x B=512 steps):
//   Phase 0: T = softmax(T_raw) stored f16, per-symbol COLUMN-major blocks
//   Phase 1: per-chunk transposed products P_g = (T_{x_{t0}}...T_{x_{t0+B-1}})^T
//            via P <- T_t^T x P.  A = T^T from TDM-staged LDS; B = P kept in
//            registers; D->B conversion in-register (cvt + v_permlanex16).
//   Phase 2: serial vector scan over chunks -> entry states
//   Phase 3: per-chunk state replay -> inc_p/dec_p
//   Phase 4: per-chunk counter transforms N_g (sparse row updates)
//   Phase 5: serial vector scan -> entry dists
//   Phase 6: per-chunk counter replay fused with output matmul + softmax

#define S_DIM 64
#define I_DIM 64
#define O_DIM 32
#define C_DIM 64
#define L_SEQ 524288
#define G_CHUNKS 1024
#define B_STEPS 512

typedef __attribute__((ext_vector_type(16))) _Float16 v16h;
typedef __attribute__((ext_vector_type(8)))  float    v8f;
typedef __attribute__((ext_vector_type(2)))  _Float16 h2;

#if __has_builtin(__builtin_amdgcn_tensor_load_to_lds)
#define USE_TDM 1
#else
#define USE_TDM 0
#endif

// Lane-half swap (lane i <-> lane i^16) without touching the LDS pipe:
// v_permlanex16_b32 with identity nibble selects reads the opposite half.
__device__ __forceinline__ unsigned half_swap(unsigned x) {
#if __has_builtin(__builtin_amdgcn_permlanex16)
  return (unsigned)__builtin_amdgcn_permlanex16(
      (int)x, (int)x, 0x76543210, 0xfedcba98, false, false);
#else
  return (unsigned)__shfl_xor((int)x, 16);
#endif
}

#if USE_TDM
typedef unsigned int v4u  __attribute__((ext_vector_type(4)));
typedef int          v4i_ __attribute__((ext_vector_type(4)));
typedef int          v8i_ __attribute__((ext_vector_type(8)));

// Per ISA 10.2: generic LDS addresses carry the LDS offset in the low 32 bits.
__device__ __forceinline__ unsigned lds_off_u32(const void* p) {
  return (unsigned)(unsigned long long)p;
}

// Issue a TDM copy of one 64x64 f16 matrix (8KB, contiguous) into LDS.
// D# group0: count=1 | lds_addr | global_addr | type=2 ("image")
// D# group1: data_size=2B, tensor_dim0=64, tensor_dim1=64, tile 64x64,
//            tensor_dim0_stride=64, no padding / multicast / iterate.
// 6-arg builtin: (uint32x4 g0, int32x8 g1, int32x4, int32x4, int32x8, cpol)
__device__ __forceinline__ void tdm_load_mat(const _Float16* T16, int k,
                                             unsigned lds_byte_off) {
  const unsigned long long ga =
      (unsigned long long)(const void*)(T16 + (size_t)k * 4096);
  v4u g0;
  g0[0] = 1u;                                             // count=1 (valid)
  g0[1] = lds_byte_off;                                   // lds_addr
  g0[2] = (unsigned)ga;                                   // global_addr[31:0]
  g0[3] = (unsigned)((ga >> 32) & 0x01FFFFFFull)          // global_addr[56:32]
          | (2u << 30);                                   // type=2
  v8i_ g1;
  g1[0] = 1 << 16;     // workgroup_mask=0 (not in cluster), data_size=1 (2B)
  g1[1] = 64 << 16;    // tensor_dim0=64 (low 16 of bits 79:48)
  g1[2] = 64 << 16;    // tensor_dim0 hi=0 | tensor_dim1=64 (low 16)
  g1[3] = 64 << 16;    // tensor_dim1 hi=0 | tile_dim0=64
  g1[4] = 64;          // tile_dim1=64 | tile_dim2=0
  g1[5] = 64;          // tensor_dim0_stride[31:0]=64
  g1[6] = 0;           // stride hi | tensor_dim1_stride lo
  g1[7] = 0;
  const v4i_ z4 = {0, 0, 0, 0};
  const v8i_ z8 = {0, 0, 0, 0, 0, 0, 0, 0};
  __builtin_amdgcn_tensor_load_to_lds(g0, g1, z4, z4, z8, 0);
}
#endif

// ---------------------------------------------------------------------------
// Phase 0a: T = softmax(T_raw, axis=-1), stored f16, per symbol COLUMN-major:
//   T16[(i*64 + col)*64 + row] = softmax_row(T_raw[row, i, :])[col]
// Column-major makes the TDM staging contiguous and both WMMA operand
// fragment reads contiguous in LDS.
__global__ void k_softmax_T(const float* __restrict__ Traw,
                            _Float16* __restrict__ T16) {
  int r = blockIdx.x;              // 0..4095 = s*64 + i
  int s = r >> 6, i = r & 63;
  int tid = threadIdx.x;           // 64 threads: one per destination column
  float v = Traw[(size_t)s * 4096 + (size_t)i * 64 + tid];
  __shared__ float red[64];
  red[tid] = v; __syncthreads();
  for (int o = 32; o > 0; o >>= 1) {
    if (tid < o) red[tid] = fmaxf(red[tid], red[tid + o]);
    __syncthreads();
  }
  float m = red[0]; __syncthreads();
  float e = __expf(v - m);
  red[tid] = e; __syncthreads();
  for (int o = 32; o > 0; o >>= 1) {
    if (tid < o) red[tid] += red[tid + o];
    __syncthreads();
  }
  // transposed store: column tid, row s of symbol i
  T16[((size_t)i * 64 + tid) * 64 + s] = (_Float16)(e / red[0]);
}

// Phase 0b: state0 = softmax(init)
__global__ void k_softmax_init(const float* __restrict__ init,
                               float* __restrict__ state0) {
  int tid = threadIdx.x;
  float v = init[tid];
  __shared__ float red[64];
  red[tid] = v; __syncthreads();
  for (int o = 32; o > 0; o >>= 1) {
    if (tid < o) red[tid] = fmaxf(red[tid], red[tid + o]);
    __syncthreads();
  }
  float m = red[0]; __syncthreads();
  float e = __expf(v - m);
  red[tid] = e; __syncthreads();
  for (int o = 32; o > 0; o >>= 1) {
    if (tid < o) red[tid] += red[tid + o];
    __syncthreads();
  }
  state0[tid] = e / red[0];
}

// ---------------------------------------------------------------------------
// Phase 1: transposed chunk products P_g = M_g^T where
// M_g = T_{x_{t0}} * ... * T_{x_{t0+B-1}} (row-vector convention).
// Recurrence: P <- T_t^T * P, ascending t.
//   - 4 waves; wave w owns P's column block [16w, 16w+16) as WMMA B operand
//     fragments B0 (K=0..31) and B1 (K=32..63), held in registers.
//   - A = T_t^T fragments, read from TDM-staged column-major LDS copy of T_k
//     (A[m][k] = T[k][m] = ldsT[m*64+k], contiguous -> ds_load_b128).
//   - D(f32, column-per-lane) -> next B(f16, column-per-lane) conversion is
//     in-register: pack row-pairs to h2; v_permlanex16 swaps lane-halves.
__global__ void __launch_bounds__(128)
k_chunk_products(const int* __restrict__ seq,
                 const _Float16* __restrict__ T16,
                 float* __restrict__ Pout) {
  __shared__ __align__(16) _Float16 ldsT[2][64 * 64];   // column-major [n*64+k]
  const int tid  = threadIdx.x;
  const int lane = tid & 31;
  const int wav  = tid >> 5;
  const int g    = blockIdx.x;

  const int half = lane >> 4;       // 0: lanes 0-15, 1: lanes 16-31
  const int ml   = lane & 15;
  const int col  = wav * 16 + ml;   // global P column owned by this lane

  // B operand state: P column block, identity-initialized.
  // B layout: lanes 0-15 hold K=0..15 (2 per VGPR), lanes 16-31 K=16..31;
  // column N = lane%16 (+ wave block).  B1 covers K+32.
  union VB { v16h v; unsigned u[8]; };
  VB b0, b1;
  #pragma unroll
  for (int v = 0; v < 8; ++v) {
    #pragma unroll
    for (int h = 0; h < 2; ++h) {
      const int k0 = 16 * half + 2 * v + h;
      b0.v[2 * v + h] = (_Float16)((k0      == col) ? 1.0f : 0.0f);
      b1.v[2 * v + h] = (_Float16)((k0 + 32 == col) ? 1.0f : 0.0f);
    }
  }

#if USE_TDM
  {
    const int k0 = __builtin_amdgcn_readfirstlane(seq[g * B_STEPS]);
    if (wav == 0) {
      tdm_load_mat(T16, k0, lds_off_u32(&ldsT[0][0]));
      __builtin_amdgcn_s_wait_tensorcnt(0);
    }
  }
  __syncthreads();
#else
  {
    const int k0 = seq[g * B_STEPS];
    const uint4* src = (const uint4*)(T16 + (size_t)k0 * 4096);
    uint4* dst = (uint4*)&ldsT[0][0];
    #pragma unroll
    for (int q = 0; q < 4; ++q) dst[tid + 128 * q] = src[tid + 128 * q];
  }
  __syncthreads();
#endif

  for (int t = 0; t < B_STEPS; ++t) {
    const int cur = t & 1;
    const _Float16* Tb = &ldsT[cur][0];

#if USE_TDM
    // DMA the next symbol's matrix into the other buffer while computing.
    if (t + 1 < B_STEPS) {
      const int kn = __builtin_amdgcn_readfirstlane(seq[g * B_STEPS + t + 1]);
      if (wav == 0) tdm_load_mat(T16, kn, lds_off_u32(&ldsT[cur ^ 1][0]));
    }
#endif

    // Four output row-tiles (i = row block of P' = column block of M'):
    // D_i = A(i,0) x B0 + A(i,1) x B1,  A(i,kt)[m][k] = T[kt*32+k][i*16+m].
    unsigned qs[4][4];   // packed h2 row-pairs of each D tile, per lane
    #pragma unroll
    for (int i = 0; i < 4; ++i) {
      // A fragment rows: m = i*16 + ml; f16 A 16x32 layout:
      // VGPR v -> K=16*half+2v (+h in pair), matching B.
      const _Float16* Arow = Tb + (i * 16 + ml) * 64;
      v16h a0 = {}, a1 = {};
      #pragma unroll
      for (int v = 0; v < 8; ++v) {
        const int kb = 16 * half + 2 * v;
        h2 p0 = *(const h2*)&Arow[kb];        // K in [0,32)
        h2 p1 = *(const h2*)&Arow[32 + kb];   // K in [32,64)
        a0[2 * v] = p0[0]; a0[2 * v + 1] = p0[1];
        a1[2 * v] = p1[0]; a1[2 * v + 1] = p1[1];
      }
      v8f c = {};
      c = __builtin_amdgcn_wmma_f32_16x16x32_f16(false, a0, false, b0.v,
                                                 (short)0, c, false, false);
      c = __builtin_amdgcn_wmma_f32_16x16x32_f16(false, a1, false, b1.v,
                                                 (short)0, c, false, false);
      // Pack D rows (16i + 8*half + 2r, +1) at this lane's column into h2.
      #pragma unroll
      for (int r = 0; r < 4; ++r) {
        union { h2 h; unsigned u; } cv;
        cv.h[0] = (_Float16)c[2 * r];
        cv.h[1] = (_Float16)c[2 * r + 1];
        qs[i][r] = cv.u;
      }
    }

    // Rebuild B0/B1 from the four packed D tiles: lane-half swap supplies the
    // other 8 rows of this lane's column (pure VALU, no LDS-pipe pressure).
    #pragma unroll
    for (int r = 0; r < 4; ++r) {
      const unsigned o0 = half_swap(qs[0][r]);
      const unsigned o1 = half_swap(qs[1][r]);
      const unsigned o2 = half_swap(qs[2][r]);
      const unsigned o3 = half_swap(qs[3][r]);
      b0.u[r]     = half ? o1       : qs[0][r];   // K 0-7   | 16-23
      b0.u[r + 4] = half ? qs[1][r] : o0;         // K 8-15  | 24-31
      b1.u[r]     = half ? o3       : qs[2][r];   // K 32-39 | 48-55
      b1.u[r + 4] = half ? qs[3][r] : o2;         // K 40-47 | 56-63
    }

#if USE_TDM
    if (wav == 0) __builtin_amdgcn_s_wait_tensorcnt(0);
    __syncthreads();   // publish next T buffer
#else
    if (t + 1 < B_STEPS) {
      const int kn = seq[g * B_STEPS + t + 1];
      const uint4* src = (const uint4*)(T16 + (size_t)kn * 4096);
      uint4* dst = (uint4*)&ldsT[cur ^ 1][0];
      __builtin_prefetch(T16 + (size_t)kn * 4096, 0, 0);
      #pragma unroll
      for (int q = 0; q < 4; ++q) dst[tid + 128 * q] = src[tid + 128 * q];
    }
    __syncthreads();
#endif
  }

  // Write out P_g (row-major, P[k][n] = M_g[n][k]) from the B registers.
  #pragma unroll
  for (int v = 0; v < 8; ++v) {
    #pragma unroll
    for (int h = 0; h < 2; ++h) {
      const int k0 = 16 * half + 2 * v + h;
      Pout[(size_t)g * 4096 + (size_t)k0 * 64 + col]        = (float)b0.v[2 * v + h];
      Pout[(size_t)g * 4096 + (size_t)(k0 + 32) * 64 + col] = (float)b1.v[2 * v + h];
    }
  }
}

// ---------------------------------------------------------------------------
// Phase 2: serial scan over chunk matrices: s'[j] = sum_i s[i] * M_g[i][j]
//        = sum_i s[i] * P_g[j][i]   (P rows contiguous per thread)
__global__ void k_scan_states(const float* __restrict__ state0,
                              const float* __restrict__ P,
                              float* __restrict__ entryS) {
  __shared__ float s[64];
  const int tid = threadIdx.x;
  s[tid] = state0[tid];
  __syncthreads();
  for (int g = 0; g < G_CHUNKS; ++g) {
    entryS[g * 64 + tid] = s[tid];
    float acc = 0.f;
    const float* Pg = P + (size_t)g * 4096 + (size_t)tid * 64;
    #pragma unroll 8
    for (int i = 0; i < 64; ++i) acc += s[i] * Pg[i];
    __syncthreads();
    s[tid] = acc;
    __syncthreads();
  }
}

// ---------------------------------------------------------------------------
// Phase 3: per-chunk state replay; emit inc_p/dec_p (probs BEFORE update).
// T16 is column-major per symbol: T_k[i][j] = T16[(k*64 + j)*64 + i], so each
// thread (j = tid) reads a contiguous 128B run per step.
__global__ void __launch_bounds__(64)
k_replay_states(const int* __restrict__ seq,
                const _Float16* __restrict__ T16,
                const float* __restrict__ incR,
                const float* __restrict__ decR,
                const float* __restrict__ entryS,
                float* __restrict__ incP,
                float* __restrict__ decP) {
  __shared__ float sb[2][64];
  __shared__ float rbuf[4];
  const int tid = threadIdx.x, lane = tid & 31, wav = tid >> 5;
  const int g = blockIdx.x;
  sb[0][tid] = entryS[g * 64 + tid];
  __syncthreads();
  int cur = 0;
  for (int t = 0; t < B_STEPS; ++t) {
    const int tt  = g * B_STEPS + t;
    const int inp = seq[tt];
    const float sv = sb[cur][tid];
    float pi = sv * incR[tid * 64 + inp];   // inc_raw[s, inp], stride I
    float pd = sv * decR[tid * 64 + inp];
    #pragma unroll
    for (int o = 16; o > 0; o >>= 1) {
      pi += __shfl_xor(pi, o);
      pd += __shfl_xor(pd, o);
    }
    if (lane == 0) { rbuf[wav * 2] = pi; rbuf[wav * 2 + 1] = pd; }
    __syncthreads();
    if (tid == 0) {
      const float il = rbuf[0] + rbuf[2];
      const float dl = rbuf[1] + rbuf[3];
      const float m  = fmaxf(fmaxf(il, dl), 0.f);
      const float ei = __expf(il - m), ed = __expf(dl - m), ez = __expf(-m);
      const float inv = 1.f / (ei + ed + ez);
      incP[tt] = ei * inv;
      decP[tt] = ed * inv;
    }
    // new_state[j] = sum_i s[i] * T_k[i][j]  (column j contiguous)
    float acc = 0.f;
    const _Float16* Tc = T16 + ((size_t)inp * 64 + tid) * 64;
    #pragma unroll 8
    for (int i = 0; i < 64; ++i) acc += sb[cur][i] * (float)Tc[i];
    __syncthreads();
    sb[cur ^ 1][tid] = acc;
    cur ^= 1;
    __syncthreads();
  }
}

// ---------------------------------------------------------------------------
// Phase 4: per-chunk counter transforms N_g (dense 64x64) via sparse updates
// N <- A_t * N, where newRow[i] = inc*Row[i-1 mod 64] + dec*Row[i+1](i<63)
//                                + noop*Row[i] + (i==0 ? dec*Row[0] : 0)
__global__ void __launch_bounds__(64)
k_counter_chunks(const float* __restrict__ incP,
                 const float* __restrict__ decP,
                 float* __restrict__ Nout) {
  __shared__ float buf[2][64 * 65];   // row-padded to 65 to dodge bank clash
  const int tid = threadIdx.x, g = blockIdx.x;
  for (int j = 0; j < 64; ++j) buf[0][tid * 65 + j] = (tid == j) ? 1.f : 0.f;
  __syncthreads();
  int cur = 0;
  for (int t = 0; t < B_STEPS; ++t) {
    const int tt = g * B_STEPS + t;
    const float inc  = incP[tt];
    const float dec  = decP[tt];
    const float noop = fmaxf(0.f, 1.f - inc - dec);
    const float* up = &buf[cur][((tid + 63) & 63) * 65];
    const float* me = &buf[cur][tid * 65];
    const float* dn = &buf[cur][((tid + 1) & 63) * 65];
    float* outr = &buf[cur ^ 1][tid * 65];
    if (tid == 0) {
      #pragma unroll 8
      for (int j = 0; j < 64; ++j)
        outr[j] = inc * up[j] + dec * (dn[j] + me[j]) + noop * me[j];
    } else if (tid == 63) {
      #pragma unroll 8
      for (int j = 0; j < 64; ++j)
        outr[j] = inc * up[j] + noop * me[j];
    } else {
      #pragma unroll 8
      for (int j = 0; j < 64; ++j)
        outr[j] = inc * up[j] + dec * dn[j] + noop * me[j];
    }
    __syncthreads();
    cur ^= 1;
  }
  for (int j = 0; j < 64; ++j)
    Nout[(size_t)g * 4096 + tid * 64 + j] = buf[cur][tid * 65 + j];
}

// ---------------------------------------------------------------------------
// Phase 5: serial scan over counter transforms: d_{g+1} = N_g d_g
__global__ void k_scan_dists(const float* __restrict__ N,
                             float* __restrict__ entryD) {
  __shared__ float d[64];
  const int tid = threadIdx.x;
  d[tid] = (tid == 0) ? 1.f : 0.f;
  __syncthreads();
  for (int g = 0; g < G_CHUNKS; ++g) {
    entryD[g * 64 + tid] = d[tid];
    float acc = 0.f;
    const float* Ng = N + (size_t)g * 4096 + (size_t)tid * 64;
    #pragma unroll 8
    for (int j = 0; j < 64; ++j) acc += Ng[j] * d[j];
    __syncthreads();
    d[tid] = acc;
    __syncthreads();
  }
}

// ---------------------------------------------------------------------------
// Phase 6: per-chunk counter replay fused with output matmul + softmax.
// Emits softmax(dist @ W^T + b) for the dist BEFORE each update.
__global__ void __launch_bounds__(64)
k_output(const float* __restrict__ incP,
         const float* __restrict__ decP,
         const float* __restrict__ entryD,
         const float* __restrict__ W,
         const float* __restrict__ bvec,
         float* __restrict__ out) {
  __shared__ float Ws[32 * 64];
  __shared__ float bs[32];
  __shared__ float db[2][64];
  const int tid = threadIdx.x, g = blockIdx.x;
  for (int idx = tid; idx < 2048; idx += 64) Ws[idx] = W[idx];
  if (tid < 32) bs[tid] = bvec[tid];
  db[0][tid] = entryD[g * 64 + tid];
  __syncthreads();
  int cur = 0;
  for (int t = 0; t < B_STEPS; ++t) {
    const int tt = g * B_STEPS + t;
    if (tid < 32) {   // wave 0: logits + softmax for this step (pre-update)
      float l = bs[tid];
      #pragma unroll 8
      for (int j = 0; j < 64; ++j) l += db[cur][j] * Ws[tid * 64 + j];
      float m = l;
      #pragma unroll
      for (int o = 16; o > 0; o >>= 1) m = fmaxf(m, __shfl_xor(m, o));
      const float e = __expf(l - m);
      float ssum = e;
      #pragma unroll
      for (int o = 16; o > 0; o >>= 1) ssum += __shfl_xor(ssum, o);
      out[(size_t)tt * 32 + tid] = e / ssum;
    }
    // all 64 threads: counter update into other buffer
    const float inc  = incP[tt];
    const float dec  = decP[tt];
    const float noop = fmaxf(0.f, 1.f - inc - dec);
    float nv = inc * db[cur][(tid + 63) & 63] + noop * db[cur][tid];
    if (tid == 0)       nv += dec * (db[cur][1] + db[cur][0]);
    else if (tid < 63)  nv += dec * db[cur][tid + 1];
    db[cur ^ 1][tid] = nv;
    __syncthreads();
    cur ^= 1;
  }
}

// ---------------------------------------------------------------------------
extern "C" void kernel_launch(void* const* d_in, const int* in_sizes, int n_in,
                              void* d_out, int out_size, void* d_ws,
                              size_t ws_size, hipStream_t stream) {
  const int*   seq  = (const int*)  d_in[0];
  const float* Traw = (const float*)d_in[1];
  const float* incR = (const float*)d_in[2];
  const float* decR = (const float*)d_in[3];
  const float* W    = (const float*)d_in[4];
  const float* bvec = (const float*)d_in[5];
  const float* init = (const float*)d_in[6];
  float* out = (float*)d_out;
  (void)in_sizes; (void)n_in; (void)out_size; (void)ws_size;

  char* ws = (char*)d_ws;
  size_t off = 0;
  auto wsalloc = [&](size_t bytes) -> char* {
    char* p = ws + off;
    off += (bytes + 255) & ~(size_t)255;
    return p;
  };
  _Float16* T16  = (_Float16*)wsalloc((size_t)64 * 64 * 64 * sizeof(_Float16));
  float* state0  = (float*)wsalloc(64 * sizeof(float));
  float* P       = (float*)wsalloc((size_t)G_CHUNKS * 4096 * sizeof(float));
  float* entryS  = (float*)wsalloc((size_t)G_CHUNKS * 64 * sizeof(float));
  float* incP    = (float*)wsalloc((size_t)L_SEQ * sizeof(float));
  float* decP    = (float*)wsalloc((size_t)L_SEQ * sizeof(float));
  float* N       = (float*)wsalloc((size_t)G_CHUNKS * 4096 * sizeof(float));
  float* entryD  = (float*)wsalloc((size_t)G_CHUNKS * 64 * sizeof(float));

  k_softmax_T     <<<4096,     64,  0, stream>>>(Traw, T16);
  k_softmax_init  <<<1,        64,  0, stream>>>(init, state0);
  k_chunk_products<<<G_CHUNKS, 128, 0, stream>>>(seq, T16, P);
  k_scan_states   <<<1,        64,  0, stream>>>(state0, P, entryS);
  k_replay_states <<<G_CHUNKS, 64,  0, stream>>>(seq, T16, incR, decR, entryS,
                                                 incP, decP);
  k_counter_chunks<<<G_CHUNKS, 64,  0, stream>>>(incP, decP, N);
  k_scan_dists    <<<1,        64,  0, stream>>>(N, entryD);
  k_output        <<<G_CHUNKS, 64,  0, stream>>>(incP, decP, entryD, W, bvec,
                                                 out);
}
